// QwenImageAttention_57269093925415
// MI455X (gfx1250) — matrix operand order repaired and checked by reading, not compiled, not run on hardware
//
#include <hip/hip_runtime.h>
#include <hip/hip_bf16.h>
#include <math.h>

typedef __attribute__((ext_vector_type(16))) _Float16 v16h;
typedef __attribute__((ext_vector_type(8)))  _Float16 v8h;
typedef __attribute__((ext_vector_type(4)))  _Float16 v4h;
typedef __attribute__((ext_vector_type(8)))  float    v8f;
typedef __attribute__((__vector_size__(4 * sizeof(int)))) int i32x4;

#define HIDDEN 2048
#define HEADS  16
#define HEAD_DIM 128
#define SEQ 2048
#define BATCH 2
#define NTOK (BATCH * SEQ)   // 4096

// log2(e)/sqrt(HEAD_DIM): folded into Q's LayerNorm so softmax runs in the
// log2 domain (bare v_exp_f32, no per-score multiplies).
#define QSCALE 0.12751744f

union V16H { v16h v; v8h h[2]; };

// ---- async global->LDS path (guarded; falls back to ds_store staging) -----
#if defined(__has_builtin)
#if __has_builtin(__builtin_amdgcn_global_load_async_to_lds_b128)
#define USE_ASYNC_LDS 1
#else
#define USE_ASYNC_LDS 0
#endif
#else
#define USE_ASYNC_LDS 0
#endif

#if USE_ASYNC_LDS
// Param types per hipcc diagnostic: (int4 AS1*, int4 AS3*, imm int, imm int)
typedef __attribute__((address_space(1))) i32x4 g_i32x4;
typedef __attribute__((address_space(3))) i32x4 l_i32x4;
#define ASYNC_CP16(gp, lp) \
    __builtin_amdgcn_global_load_async_to_lds_b128((g_i32x4*)(gp), (l_i32x4*)(lp), 0, 0)
#endif

__device__ __forceinline__ void wait_async0() {
#if defined(__has_builtin)
#if __has_builtin(__builtin_amdgcn_s_wait_asynccnt)
    __builtin_amdgcn_s_wait_asynccnt(0);
#else
    asm volatile("s_wait_asynccnt 0x0" ::: "memory");
#endif
#else
    asm volatile("s_wait_asynccnt 0x0" ::: "memory");
#endif
}

__device__ __forceinline__ float fexp2(float x) {
#if defined(__has_builtin)
#if __has_builtin(__builtin_amdgcn_exp2f)
    return __builtin_amdgcn_exp2f(x);
#else
    return exp2f(x);
#endif
#else
    return exp2f(x);
#endif
}

// ---- DPP16 butterfly reductions (VALU-only, no ds_bpermute) ---------------
#if defined(__has_builtin)
#if __has_builtin(__builtin_amdgcn_update_dpp)
#define HAVE_DPP 1
#else
#define HAVE_DPP 0
#endif
#else
#define HAVE_DPP 0
#endif

#if HAVE_DPP
template<int CTRL>
__device__ __forceinline__ float dpp_bfly(float x) {
    return __int_as_float(
        __builtin_amdgcn_update_dpp(0, __float_as_int(x), CTRL, 0xF, 0xF, true));
}
#endif

// max/sum over each 16-lane half-wave (butterfly; valid for reductions)
__device__ __forceinline__ float redmax16(float v) {
#if HAVE_DPP
    v = fmaxf(v, dpp_bfly<0xB1>(v));    // quad_perm(1,0,3,2)  ~ xor 1
    v = fmaxf(v, dpp_bfly<0x4E>(v));    // quad_perm(2,3,0,1)  ~ xor 2
    v = fmaxf(v, dpp_bfly<0x141>(v));   // row_half_mirror     ~ xor 4
    v = fmaxf(v, dpp_bfly<0x140>(v));   // row_mirror          ~ xor 8
#else
#pragma unroll
    for (int off = 1; off < 16; off <<= 1) v = fmaxf(v, __shfl_xor(v, off, 32));
#endif
    return v;
}

__device__ __forceinline__ float redsum16(float v) {
#if HAVE_DPP
    v += dpp_bfly<0xB1>(v);
    v += dpp_bfly<0x4E>(v);
    v += dpp_bfly<0x141>(v);
    v += dpp_bfly<0x140>(v);
#else
#pragma unroll
    for (int off = 1; off < 16; off <<= 1) v += __shfl_xor(v, off, 32);
#endif
    return v;
}

// sum over all 32 lanes (DPP within rows + one cross-row shuffle)
__device__ __forceinline__ float redsum32(float v) {
    v = redsum16(v);
    v += __shfl_xor(v, 16, 32);
    return v;
}

// ---------------------------------------------------------------------------
// fp32 -> f16 elementwise convert (4 elems/thread)
// ---------------------------------------------------------------------------
__global__ __launch_bounds__(256)
void convert_f32_to_f16(const float* __restrict__ in, _Float16* __restrict__ out) {
    size_t i = ((size_t)blockIdx.x * 256 + threadIdx.x) * 4;
    float4 f = *(const float4*)(in + i);
    v4h h;
    h[0] = (_Float16)f.x; h[1] = (_Float16)f.y;
    h[2] = (_Float16)f.z; h[3] = (_Float16)f.w;
    *(v4h*)(out + i) = h;
}

// ---------------------------------------------------------------------------
// fp32 [K,N] -> f16 transposed [N,K] (LDS-tiled for coalescing)
// ---------------------------------------------------------------------------
__global__ __launch_bounds__(256)
void transpose_f32_to_f16(const float* __restrict__ in, _Float16* __restrict__ out,
                          int K, int N) {
    __shared__ float tile[32][33];
    int x = blockIdx.x * 32 + threadIdx.x;
    int y = blockIdx.y * 32 + threadIdx.y;
#pragma unroll
    for (int j = 0; j < 32; j += 8)
        tile[threadIdx.y + j][threadIdx.x] = in[(size_t)(y + j) * N + x];
    __syncthreads();
    int x2 = blockIdx.y * 32 + threadIdx.x;
    int y2 = blockIdx.x * 32 + threadIdx.y;
#pragma unroll
    for (int j = 0; j < 32; j += 8)
        out[(size_t)(y2 + j) * K + x2] = (_Float16)tile[threadIdx.x][threadIdx.y + j];
}

// ---------------------------------------------------------------------------
// Tiled WMMA GEMM: C[M,N] = A[M,K](f16 row-major) * Bt[N,K](f16) + bias[n].
// MODE: 0 fp32 [M,N] (O-proj -> d_out); 1 f16 [B,H,S,D]; 2 f16 [B,H,D,S].
// 8 waves, tile 128x128, wave tile 64x32, K-step 32.
// ---------------------------------------------------------------------------
template<int MODE>
__global__ __launch_bounds__(256)
void gemm16(const _Float16* __restrict__ A, const _Float16* __restrict__ Bt,
            const float* __restrict__ bias, void* __restrict__ Out,
            int M, int N, int K) {
    const int tid  = threadIdx.x;
    const int wave = tid >> 5;
    const int lane = tid & 31;
    const int wm = wave >> 2;
    const int wn = wave & 3;
    const int m0 = blockIdx.x * 128 + wm * 64;
    const int n0 = blockIdx.y * 128 + wn * 32;

    const int am = lane & 15;
    const int ak = (lane & 16) ? 16 : 0;
    const int bn = lane & 15;
    const int bk = (lane & 16) ? 16 : 0;
    const int half8 = (lane & 16) ? 8 : 0;

    v8f acc[4][2];
#pragma unroll
    for (int i = 0; i < 4; ++i)
#pragma unroll
        for (int j = 0; j < 2; ++j)
            acc[i][j] = (v8f)(0.0f);

    for (int k0 = 0; k0 < K; k0 += 32) {
        v16h a[4];
#pragma unroll
        for (int i = 0; i < 4; ++i) {
            const _Float16* p = A + (size_t)(m0 + i * 16 + am) * K + k0 + ak;
            V16H t;
            t.h[0] = *(const v8h*)p;
            t.h[1] = *(const v8h*)(p + 8);
            a[i] = t.v;
        }
        v16h b[2];
#pragma unroll
        for (int j = 0; j < 2; ++j)
            b[j] = *(const v16h*)(Bt + (size_t)(n0 + j * 16 + bn) * K + k0 + bk);
#pragma unroll
        for (int i = 0; i < 4; ++i)
#pragma unroll
            for (int j = 0; j < 2; ++j)
                acc[i][j] = __builtin_amdgcn_wmma_f32_16x16x32_f16(
                    false, a[i], false, b[j], (short)0, acc[i][j], false, false);
    }

#pragma unroll
    for (int i = 0; i < 4; ++i) {
#pragma unroll
        for (int j = 0; j < 2; ++j) {
            const int n = n0 + j * 16 + (lane & 15);
            const float bv = bias[n];
#pragma unroll
            for (int r = 0; r < 8; ++r) {
                const int m = m0 + i * 16 + r + half8;
                const float c = acc[i][j][r] + bv;
                if (MODE == 0) {
                    ((float*)Out)[(size_t)m * N + n] = c;
                } else {
                    const int bb = m >> 11;
                    const int s  = m & 2047;
                    const int h  = n >> 7;
                    const int d  = n & 127;
                    size_t idx;
                    if (MODE == 1)
                        idx = ((size_t)(bb * HEADS + h) * SEQ + s) * HEAD_DIM + d;
                    else
                        idx = ((size_t)(bb * HEADS + h) * HEAD_DIM + d) * SEQ + s;
                    ((_Float16*)Out)[idx] = (_Float16)c;
                }
            }
        }
    }
}

// ---------------------------------------------------------------------------
// Per-head LayerNorm over 128 dims, in-place on f16 [B,H,S,D].
// `scale` is folded into gamma AND beta (used to pre-scale Q for softmax).
// ---------------------------------------------------------------------------
__global__ __launch_bounds__(256)
void ln_head(_Float16* __restrict__ X, const float* __restrict__ gamma,
             const float* __restrict__ beta, float scale) {
    const int row  = blockIdx.x * 8 + (threadIdx.x >> 5);
    const int lane = threadIdx.x & 31;
    _Float16* p = X + (size_t)row * HEAD_DIM + lane * 4;
    v4h d = *(v4h*)p;
    float x[4];
#pragma unroll
    for (int e = 0; e < 4; ++e) x[e] = (float)d[e];
    float s = redsum32(x[0] + x[1] + x[2] + x[3]);
    const float mean = s * (1.0f / 128.0f);
    float vs = 0.f;
#pragma unroll
    for (int e = 0; e < 4; ++e) { float t = x[e] - mean; vs += t * t; }
    vs = redsum32(vs);
    const float inv = rsqrtf(vs * (1.0f / 128.0f) + 1e-6f);
#pragma unroll
    for (int e = 0; e < 4; ++e) {
        const int c = lane * 4 + e;
        d[e] = (_Float16)(((x[e] - mean) * inv * gamma[c] + beta[c]) * scale);
    }
    *(v4h*)p = d;
}

// ---------------------------------------------------------------------------
// Flash attention with double-buffered LDS staging of K/V tiles.
// Qn/Kn: f16 [B,H,S,D]; Vt: f16 [B,H,D,S]; Out: f16 [NTOK, HIDDEN].
// Q is pre-scaled by log2(e)/sqrt(D) -> softmax uses exp2.
// Grid (S/128, B*H), 256 threads = 8 waves x 16 query rows.
// ---------------------------------------------------------------------------
struct Stage { v8h k0, k1, va, vb_; };

__device__ __forceinline__ void stage_issue(const _Float16* __restrict__ Kt,
                                            const _Float16* __restrict__ V,
                                            int j0, int tid,
                                            _Float16 (*kb)[128], _Float16 (*vb)[32],
                                            Stage& st) {
    const _Float16* gk = Kt + (size_t)(j0 + (tid >> 3)) * HEAD_DIM + (tid & 7) * 16;
    const _Float16* gv = V  + (size_t)(tid >> 1) * SEQ + j0 + (tid & 1) * 16;
    _Float16* lk = &kb[tid >> 3][(tid & 7) * 16];
    _Float16* lv = &vb[tid >> 1][(tid & 1) * 16];
#if USE_ASYNC_LDS
    ASYNC_CP16(gk,     lk);
    ASYNC_CP16(gk + 8, lk + 8);
    ASYNC_CP16(gv,     lv);
    ASYNC_CP16(gv + 8, lv + 8);
    (void)st;
#else
    st.k0  = *(const v8h*)gk;
    st.k1  = *(const v8h*)(gk + 8);
    st.va  = *(const v8h*)gv;
    st.vb_ = *(const v8h*)(gv + 8);
    (void)lk; (void)lv;
#endif
}

__device__ __forceinline__ void stage_commit(int tid, _Float16 (*kb)[128],
                                             _Float16 (*vb)[32], const Stage& st) {
#if !USE_ASYNC_LDS
    *(v8h*)&kb[tid >> 3][(tid & 7) * 16]     = st.k0;
    *(v8h*)&kb[tid >> 3][(tid & 7) * 16 + 8] = st.k1;
    *(v8h*)&vb[tid >> 1][(tid & 1) * 16]     = st.va;
    *(v8h*)&vb[tid >> 1][(tid & 1) * 16 + 8] = st.vb_;
#else
    (void)tid; (void)kb; (void)vb; (void)st;
#endif
}

__global__ __launch_bounds__(256)
void attn_fa(const _Float16* __restrict__ Qn, const _Float16* __restrict__ Kn,
             const _Float16* __restrict__ Vt, _Float16* __restrict__ Out) {
    __shared__ alignas(32) _Float16 kbuf[2][32][128];  // 2 x 8 KB, key rows
    __shared__ alignas(32) _Float16 vbuf[2][128][32];  // 2 x 8 KB, [d][s]
    __shared__ alignas(32) _Float16 pbuf[8][16][32];   // per-wave P relayout

    const int bh = blockIdx.y;
    const _Float16* Q  = Qn + (size_t)bh * SEQ * HEAD_DIM;
    const _Float16* Kt = Kn + (size_t)bh * SEQ * HEAD_DIM;
    const _Float16* V  = Vt + (size_t)bh * HEAD_DIM * SEQ;

    const int tid  = threadIdx.x;
    const int wave = tid >> 5;
    const int lane = tid & 31;
    const int m0 = blockIdx.x * 128 + wave * 16;

    const int am = lane & 15;
    const int ak = (lane & 16) ? 16 : 0;
    const int bn = lane & 15;
    const int bk = (lane & 16) ? 16 : 0;
    const int half8 = (lane & 16) ? 8 : 0;

    // Resident Q fragments (already scaled by log2(e)/sqrt(D))
    v16h qf[4];
#pragma unroll
    for (int c = 0; c < 4; ++c) {
        const _Float16* p = Q + (size_t)(m0 + am) * HEAD_DIM + c * 32 + ak;
        V16H t; t.h[0] = *(const v8h*)p; t.h[1] = *(const v8h*)(p + 8);
        qf[c] = t.v;
    }

    v8f oacc[8];
#pragma unroll
    for (int t = 0; t < 8; ++t) oacc[t] = (v8f)(0.0f);
    float mrow[8], lrow[8];
#pragma unroll
    for (int r = 0; r < 8; ++r) { mrow[r] = -INFINITY; lrow[r] = 0.f; }

    // Prologue: stage block 0
    Stage st;
    stage_issue(Kt, V, 0, tid, kbuf[0], vbuf[0], st);
    stage_commit(tid, kbuf[0], vbuf[0], st);
#if USE_ASYNC_LDS
    wait_async0();
#endif
    __syncthreads();

    const int NBLK = SEQ / 32;
    for (int i = 0; i < NBLK; ++i) {
        const int cur = i & 1, nxt = cur ^ 1;
        if (i + 1 < NBLK)
            stage_issue(Kt, V, (i + 1) * 32, tid, kbuf[nxt], vbuf[nxt], st);

        // ---- scores (log2 domain): two 16x16 tiles ----
        v8f sfr[2];
        sfr[0] = (v8f)(0.0f); sfr[1] = (v8f)(0.0f);
#pragma unroll
        for (int c = 0; c < 4; ++c) {
#pragma unroll
            for (int jj = 0; jj < 2; ++jj) {
                v16h bf = *(const v16h*)&kbuf[cur][jj * 16 + bn][c * 32 + bk];
                sfr[jj] = __builtin_amdgcn_wmma_f32_16x16x32_f16(
                    false, qf[c], false, bf, (short)0, sfr[jj], false, false);
            }
        }
        // ---- online softmax (exp2; DPP16 half-wave row reductions) ----
        float corr[8];
#pragma unroll
        for (int r = 0; r < 8; ++r) {
            const float s0 = sfr[0][r], s1 = sfr[1][r];
            const float mx = redmax16(fmaxf(s0, s1));
            const float newm = fmaxf(mrow[r], mx);
            corr[r] = fexp2(mrow[r] - newm);
            mrow[r] = newm;
            const float p0 = fexp2(s0 - newm);
            const float p1 = fexp2(s1 - newm);
            sfr[0][r] = p0; sfr[1][r] = p1;
            const float ps = redsum16(p0 + p1);
            lrow[r] = lrow[r] * corr[r] + ps;
        }
#pragma unroll
        for (int t = 0; t < 8; ++t)
#pragma unroll
            for (int r = 0; r < 8; ++r)
                oacc[t][r] *= corr[r];

        // ---- relayout P (C-frag) -> A-frag via wave-private LDS ----
#pragma unroll
        for (int jj = 0; jj < 2; ++jj)
#pragma unroll
            for (int r = 0; r < 8; ++r)
                pbuf[wave][r + half8][jj * 16 + bn] = (_Float16)sfr[jj][r];
        const _Float16* pr = &pbuf[wave][am][ak];
        V16H pt; pt.h[0] = *(const v8h*)pr; pt.h[1] = *(const v8h*)(pr + 8);
        v16h pf = pt.v;

        // ---- O += P * V ----
#pragma unroll
        for (int t = 0; t < 8; ++t) {
            v16h bf = *(const v16h*)&vbuf[cur][t * 16 + bn][bk];
            oacc[t] = __builtin_amdgcn_wmma_f32_16x16x32_f16(
                false, pf, false, bf, (short)0, oacc[t], false, false);
        }

        if (i + 1 < NBLK)
            stage_commit(tid, kbuf[nxt], vbuf[nxt], st);
#if USE_ASYNC_LDS
        wait_async0();
#endif
        __syncthreads();
    }

    // ---- epilogue: O /= l, write f16 token-major [NTOK, HIDDEN] ----
    const int bb = bh >> 4;
    const int h  = bh & 15;
#pragma unroll
    for (int t = 0; t < 8; ++t) {
#pragma unroll
        for (int r = 0; r < 8; ++r) {
            const float o = oacc[t][r] / lrow[r];
            const int srow = m0 + r + half8;
            const size_t token = (size_t)bb * SEQ + srow;
            const int col = h * HEAD_DIM + t * 16 + bn;
            Out[token * HIDDEN + col] = (_Float16)o;
        }
    }
}

// ---------------------------------------------------------------------------
// Host launcher
// ---------------------------------------------------------------------------
extern "C" void kernel_launch(void* const* d_in, const int* in_sizes, int n_in,
                              void* d_out, int out_size, void* d_ws, size_t ws_size,
                              hipStream_t stream) {
    const float* hidden = (const float*)d_in[0];
    const float* q_w = (const float*)d_in[1];
    const float* q_b = (const float*)d_in[2];
    const float* k_w = (const float*)d_in[3];
    const float* k_b = (const float*)d_in[4];
    const float* v_w = (const float*)d_in[5];
    const float* v_b = (const float*)d_in[6];
    const float* o_w = (const float*)d_in[7];
    const float* o_b = (const float*)d_in[8];
    const float* qg  = (const float*)d_in[9];
    const float* qbt = (const float*)d_in[10];
    const float* kg  = (const float*)d_in[11];
    const float* kbt = (const float*)d_in[12];

    const size_t MB = 1u << 20;
    char* ws = (char*)d_ws;
    _Float16* Xh   = (_Float16*)(ws + 0 * MB);
    _Float16* WqT  = (_Float16*)(ws + 16 * MB);
    _Float16* WkT  = (_Float16*)(ws + 24 * MB);
    _Float16* WvT  = (_Float16*)(ws + 32 * MB);
    _Float16* WoT  = (_Float16*)(ws + 40 * MB);
    _Float16* q16  = (_Float16*)(ws + 48 * MB);
    _Float16* k16  = (_Float16*)(ws + 64 * MB);
    _Float16* vT16 = (_Float16*)(ws + 80 * MB);
    _Float16* att  = (_Float16*)(ws + 96 * MB);

    convert_f32_to_f16<<<(NTOK * HIDDEN) / (256 * 4), 256, 0, stream>>>(hidden, Xh);

    dim3 tb(32, 8), tg(HIDDEN / 32, HIDDEN / 32);
    transpose_f32_to_f16<<<tg, tb, 0, stream>>>(q_w, WqT, HIDDEN, HIDDEN);
    transpose_f32_to_f16<<<tg, tb, 0, stream>>>(k_w, WkT, HIDDEN, HIDDEN);
    transpose_f32_to_f16<<<tg, tb, 0, stream>>>(v_w, WvT, HIDDEN, HIDDEN);
    transpose_f32_to_f16<<<tg, tb, 0, stream>>>(o_w, WoT, HIDDEN, HIDDEN);

    dim3 gg(NTOK / 128, HIDDEN / 128);
    gemm16<1><<<gg, 256, 0, stream>>>(Xh, WqT, q_b, q16,  NTOK, HIDDEN, HIDDEN);
    gemm16<1><<<gg, 256, 0, stream>>>(Xh, WkT, k_b, k16,  NTOK, HIDDEN, HIDDEN);
    gemm16<2><<<gg, 256, 0, stream>>>(Xh, WvT, v_b, vT16, NTOK, HIDDEN, HIDDEN);

    const int nrows = BATCH * HEADS * SEQ;
    ln_head<<<nrows / 8, 256, 0, stream>>>(q16, qg, qbt, QSCALE);
    ln_head<<<nrows / 8, 256, 0, stream>>>(k16, kg, kbt, 1.0f);

    dim3 ag(SEQ / 128, BATCH * HEADS);
    attn_fa<<<ag, 256, 0, stream>>>(q16, k16, vT16, att);

    gemm16<0><<<gg, 256, 0, stream>>>(att, WoT, o_b, d_out, NTOK, HIDDEN, HIDDEN);
}
